// GAAE_10685878632450
// MI455X (gfx1250) — matrix operand order repaired and checked by reading, not compiled, not run on hardware
//
#include <hip/hip_runtime.h>
#include <hip/hip_bf16.h>

// ---------------- types for WMMA ----------------
typedef _Float16 v16h __attribute__((ext_vector_type(16)));
typedef _Float16 v8h  __attribute__((ext_vector_type(8)));
typedef float    v8f  __attribute__((ext_vector_type(8)));

#define NEG_SLOPE 0.2f

// ---------------- small helpers ----------------
__device__ __forceinline__ float wave_reduce_add(float v) {
    #pragma unroll
    for (int m = 16; m > 0; m >>= 1) v += __shfl_xor(v, m, 32);
    return v;
}

__device__ __forceinline__ void atomicMaxF(float* addr, float val) {
    // signed-int max for >=0, unsigned-int min for <0; correct with -inf init
    if (val >= 0.0f) atomicMax((int*)addr, __float_as_int(val));
    else             atomicMin((unsigned int*)addr, __float_as_uint(val));
}

// ---------------- fill / convert kernels ----------------
__global__ void fill_f32(float* p, float v, int n) {
    int i = blockIdx.x * blockDim.x + threadIdx.x;
    if (i < n) p[i] = v;
}

__global__ void cvt_f32_to_f16(const float* in, _Float16* out, int n) {
    int i = blockIdx.x * blockDim.x + threadIdx.x;
    if (i < n) out[i] = (_Float16)in[i];
}

// W is [K, Nout] row-major f32 -> Wt is [Nout, K] row-major f16
__global__ void transpose_cvt_f16(const float* W, _Float16* Wt, int K, int Nout) {
    int i = blockIdx.x * blockDim.x + threadIdx.x;
    if (i >= K * Nout) return;
    int n = i / K, k = i - n * K;
    Wt[i] = (_Float16)W[(size_t)k * Nout + n];
}

// ---------------- WMMA GEMM: C[MxN] = A[MxK](f16,row) * Bt[NxK](f16,row) ----------------
// one wave32 per 16x16 output tile; K multiple of 32; M multiple of 16
__global__ void wmma_gemm_f16(const _Float16* __restrict__ A,
                              const _Float16* __restrict__ Bt,
                              float* __restrict__ C,
                              int M, int N, int K) {
    const int wave = blockIdx.x * (blockDim.x >> 5) + (threadIdx.x >> 5);
    const int lane = threadIdx.x & 31;
    const int tilesN = N >> 4;
    const int totalTiles = (M >> 4) * tilesN;
    if (wave >= totalTiles) return;
    const int tM = wave / tilesN;
    const int tN = wave - tM * tilesN;
    const int sel = lane >> 4;      // 0: lanes 0-15, 1: lanes 16-31
    const int l   = lane & 15;

    const _Float16* arow = A  + (size_t)(tM * 16 + l) * K;   // A row M = l (this half)
    const _Float16* brow = Bt + (size_t)(tN * 16 + l) * K;   // B column N = l

    v8f c = {};
    #pragma unroll 4
    for (int k = 0; k < K; k += 32) {
        // A fragment layout (16-bit A 16x32): lane<16 -> K 0..7 & 16..23 ; lane>=16 -> K 8..15 & 24..31
        v8h a_lo = *(const v8h*)(arow + k + sel * 8);
        v8h a_hi = *(const v8h*)(arow + k + sel * 8 + 16);
        // B fragment layout (16-bit B 32x16): lane<16 -> K 0..7 & 16..23 ; lane>=16 -> K 8..15 & 24..31
        v8h b_lo = *(const v8h*)(brow + k + sel * 8);
        v8h b_hi = *(const v8h*)(brow + k + sel * 8 + 16);
        v16h a, b;
        #pragma unroll
        for (int i = 0; i < 8; ++i) { a[i] = a_lo[i]; a[8 + i] = a_hi[i];
                                      b[i] = b_lo[i]; b[8 + i] = b_hi[i]; }
        c = __builtin_amdgcn_wmma_f32_16x16x32_f16(false, a, false, b,
                                                   (short)0, c, false, false);
    }
    // C layout: lane<16: rows 0..7 col l ; lane>=16: rows 8..15 col l
    float* cbase = C + (size_t)(tM * 16 + sel * 8) * N + tN * 16 + l;
    #pragma unroll
    for (int r = 0; r < 8; ++r) cbase[(size_t)r * N] = c[r];
}

// ---------------- attention scalars: a[n,h] = dot(xh[n,h,:], att[h,:]) ----------------
__global__ void att_scalars(const float* __restrict__ xh,
                            const float* __restrict__ att_src,
                            const float* __restrict__ att_dst,
                            float* __restrict__ a_src, float* __restrict__ a_dst,
                            int N, int H, int C) {
    const int wave = blockIdx.x * (blockDim.x >> 5) + (threadIdx.x >> 5);
    const int lane = threadIdx.x & 31;
    if (wave >= N) return;
    const float* row = xh + (size_t)wave * H * C;
    for (int h = 0; h < H; ++h) {
        float ss = 0.f, dd = 0.f;
        for (int cc = lane; cc < C; cc += 32) {
            float v = row[h * C + cc];
            ss += v * att_src[h * C + cc];
            dd += v * att_dst[h * C + cc];
        }
        ss = wave_reduce_add(ss);
        dd = wave_reduce_add(dd);
        if (lane == 0) { a_src[wave * H + h] = ss; a_dst[wave * H + h] = dd; }
    }
}

// ---------------- edge phase (self-loops appended implicitly: i>=E -> src=dst=i-E) ----------------
__global__ void edge_max(const int* __restrict__ ei, int E, int N,
                         const float* __restrict__ a_src, const float* __restrict__ a_dst,
                         float* __restrict__ mx, int H) {
    int i = blockIdx.x * blockDim.x + threadIdx.x;
    int EP = E + N;
    if (i >= EP) return;
    int s, d;
    if (i < E) { s = ei[i]; d = ei[E + i]; } else { s = d = i - E; }
    for (int h = 0; h < H; ++h) {
        float e = a_src[s * H + h] + a_dst[d * H + h];
        e = (e > 0.f) ? e : NEG_SLOPE * e;
        atomicMaxF(&mx[d * H + h], e);
    }
}

__global__ void edge_exp(const int* __restrict__ ei, int E, int N,
                         const float* __restrict__ a_src, const float* __restrict__ a_dst,
                         const float* __restrict__ mx,
                         float* __restrict__ ex, float* __restrict__ sum, int H) {
    int i = blockIdx.x * blockDim.x + threadIdx.x;
    int EP = E + N;
    if (i >= EP) return;
    int s, d;
    if (i < E) { s = ei[i]; d = ei[E + i]; } else { s = d = i - E; }
    for (int h = 0; h < H; ++h) {
        float e = a_src[s * H + h] + a_dst[d * H + h];
        e = (e > 0.f) ? e : NEG_SLOPE * e;
        float v = __expf(e - mx[d * H + h]);
        ex[(size_t)i * H + h] = v;
        atomicAdd(&sum[d * H + h], v);
    }
}

// one wave per edge: coalesced gather of xh[src,:], scaled atomic scatter to agg[dst,:]
__global__ void edge_aggregate(const int* __restrict__ ei, int E, int N,
                               const float* __restrict__ xh,
                               const float* __restrict__ ex, const float* __restrict__ sum,
                               float* __restrict__ agg, int H, int HC, int cshift) {
    const int wave = blockIdx.x * (blockDim.x >> 5) + (threadIdx.x >> 5);
    const int lane = threadIdx.x & 31;
    const int EP = E + N;
    if (wave >= EP) return;
    int s, d;
    if (wave < E) { s = ei[wave]; d = ei[E + wave]; } else { s = d = wave - E; }
    float alpha[2];
    for (int h = 0; h < H; ++h)
        alpha[h] = ex[(size_t)wave * H + h] / (sum[d * H + h] + 1e-16f);
    const float* srow = xh + (size_t)s * HC;
    float* drow = agg + (size_t)d * HC;
    for (int cc = lane; cc < HC; cc += 32) {
        int h = cc >> cshift;
        atomicAdd(&drow[cc], srow[cc] * alpha[h]);
    }
}

// layer-1 finalize: z = relu(agg + bias) in place, plus f16 copy for next GEMM
__global__ void finalize_relu_f16(float* __restrict__ agg, const float* __restrict__ bias,
                                  _Float16* __restrict__ zh, int total, int HC) {
    int i = blockIdx.x * blockDim.x + threadIdx.x;
    if (i >= total) return;
    int c = i % HC;
    float v = agg[i] + bias[c];
    v = v > 0.f ? v : 0.f;
    agg[i] = v;
    zh[i] = (_Float16)v;
}

// layer-2 finalize: z = agg + bias (H=1, concat=False -> mean over 1 head == identity)
__global__ void finalize_bias(float* __restrict__ agg, const float* __restrict__ bias,
                              int total, int HC) {
    int i = blockIdx.x * blockDim.x + threadIdx.x;
    if (i >= total) return;
    agg[i] += bias[i % HC];
}

// decode: one wave per ORIGINAL edge, 64-dim dot
__global__ void decode_edges(const int* __restrict__ ei, int E,
                             const float* __restrict__ z, float* __restrict__ out) {
    const int wave = blockIdx.x * (blockDim.x >> 5) + (threadIdx.x >> 5);
    const int lane = threadIdx.x & 31;
    if (wave >= E) return;
    int s = ei[wave], d = ei[E + wave];
    const float* zs = z + (size_t)s * 64;
    const float* zd = z + (size_t)d * 64;
    float acc = zs[lane] * zd[lane] + zs[lane + 32] * zd[lane + 32];
    acc = wave_reduce_add(acc);
    if (lane == 0) out[wave] = acc;
}

// ---------------- host launch ----------------
extern "C" void kernel_launch(void* const* d_in, const int* in_sizes, int n_in,
                              void* d_out, int out_size, void* d_ws, size_t ws_size,
                              hipStream_t stream) {
    const float* x        = (const float*)d_in[0];
    const int*   ei       = (const int*)  d_in[1];   // [2,E] flat: src then dst
    const float* W1       = (const float*)d_in[2];   // [128, 256]
    const float* att_src1 = (const float*)d_in[3];   // [2,128]
    const float* att_dst1 = (const float*)d_in[4];
    const float* b1       = (const float*)d_in[5];   // [256]
    const float* W2       = (const float*)d_in[6];   // [256, 64]
    const float* att_src2 = (const float*)d_in[7];   // [1,64]
    const float* att_dst2 = (const float*)d_in[8];
    const float* b2       = (const float*)d_in[9];   // [64]
    float* out = (float*)d_out;

    const int IN  = 128;
    const int HC1 = 256;  // H1=2, C1=128
    const int HC2 = 64;   // H2=1, C2=64
    const int H1 = 2, C1 = 128, H2 = 1;
    const int N  = in_sizes[0] / IN;
    const int E  = in_sizes[1] / 2;
    const int EP = E + N;

    // ---- workspace sub-allocation (256B aligned) ----
    size_t off = 0;
    auto alloc = [&](size_t bytes) -> char* {
        char* p = (char*)d_ws + off;
        off = (off + bytes + 255) & ~(size_t)255;
        return p;
    };
    _Float16* x_h   = (_Float16*)alloc((size_t)N * IN  * 2);
    _Float16* W1t   = (_Float16*)alloc((size_t)HC1 * IN * 2);
    float*    xh1   = (float*)   alloc((size_t)N * HC1 * 4);
    float*    asr1  = (float*)   alloc((size_t)N * H1 * 4);
    float*    ads1  = (float*)   alloc((size_t)N * H1 * 4);
    float*    mx1   = (float*)   alloc((size_t)N * H1 * 4);
    float*    sm1   = (float*)   alloc((size_t)N * H1 * 4);
    float*    ex1   = (float*)   alloc((size_t)EP * H1 * 4);
    float*    agg1  = (float*)   alloc((size_t)N * HC1 * 4);   // becomes z1 in place
    _Float16* z1h   = (_Float16*)alloc((size_t)N * HC1 * 2);
    _Float16* W2t   = (_Float16*)alloc((size_t)HC2 * HC1 * 2);
    float*    xh2   = (float*)   alloc((size_t)N * HC2 * 4);
    float*    asr2  = (float*)   alloc((size_t)N * 4);
    float*    ads2  = (float*)   alloc((size_t)N * 4);
    float*    mx2   = (float*)   alloc((size_t)N * 4);
    float*    sm2   = (float*)   alloc((size_t)N * 4);
    float*    ex2   = (float*)   alloc((size_t)EP * 4);
    float*    agg2  = (float*)   alloc((size_t)N * HC2 * 4);   // becomes z2 in place
    (void)ws_size;

    const int TB = 256;
    auto blks  = [&](long long n) { return (int)((n + TB - 1) / TB); };
    auto wblks = [&](long long waves) { return (int)((waves + 7) / 8); }; // 8 waves/block

    // ===== layer 1 =====
    cvt_f32_to_f16 <<<blks((long long)N * IN), TB, 0, stream>>>(x, x_h, N * IN);
    transpose_cvt_f16<<<blks(HC1 * IN), TB, 0, stream>>>(W1, W1t, IN, HC1);
    wmma_gemm_f16 <<<wblks((long long)(N / 16) * (HC1 / 16)), TB, 0, stream>>>(
        x_h, W1t, xh1, N, HC1, IN);
    att_scalars   <<<wblks(N), TB, 0, stream>>>(xh1, att_src1, att_dst1, asr1, ads1, N, H1, C1);
    fill_f32      <<<blks(N * H1), TB, 0, stream>>>(mx1, -INFINITY, N * H1);
    fill_f32      <<<blks(N * H1), TB, 0, stream>>>(sm1, 0.f, N * H1);
    fill_f32      <<<blks((long long)N * HC1), TB, 0, stream>>>(agg1, 0.f, N * HC1);
    edge_max      <<<blks(EP), TB, 0, stream>>>(ei, E, N, asr1, ads1, mx1, H1);
    edge_exp      <<<blks(EP), TB, 0, stream>>>(ei, E, N, asr1, ads1, mx1, ex1, sm1, H1);
    edge_aggregate<<<wblks(EP), TB, 0, stream>>>(ei, E, N, xh1, ex1, sm1, agg1, H1, HC1, 7);
    finalize_relu_f16<<<blks((long long)N * HC1), TB, 0, stream>>>(agg1, b1, z1h, N * HC1, HC1);

    // ===== layer 2 =====
    transpose_cvt_f16<<<blks(HC2 * HC1), TB, 0, stream>>>(W2, W2t, HC1, HC2);
    wmma_gemm_f16 <<<wblks((long long)(N / 16) * (HC2 / 16)), TB, 0, stream>>>(
        z1h, W2t, xh2, N, HC2, HC1);
    att_scalars   <<<wblks(N), TB, 0, stream>>>(xh2, att_src2, att_dst2, asr2, ads2, N, H2, HC2);
    fill_f32      <<<blks(N), TB, 0, stream>>>(mx2, -INFINITY, N);
    fill_f32      <<<blks(N), TB, 0, stream>>>(sm2, 0.f, N);
    fill_f32      <<<blks((long long)N * HC2), TB, 0, stream>>>(agg2, 0.f, N * HC2);
    edge_max      <<<blks(EP), TB, 0, stream>>>(ei, E, N, asr2, ads2, mx2, H2);
    edge_exp      <<<blks(EP), TB, 0, stream>>>(ei, E, N, asr2, ads2, mx2, ex2, sm2, H2);
    edge_aggregate<<<wblks(EP), TB, 0, stream>>>(ei, E, N, xh2, ex2, sm2, agg2, H2, HC2, 6);
    finalize_bias <<<blks((long long)N * HC2), TB, 0, stream>>>(agg2, b2, N * HC2, HC2);

    // ===== decode over original edges =====
    decode_edges  <<<wblks(E), TB, 0, stream>>>(ei, E, agg2, out);
    (void)out_size; (void)n_in;
}
